// CrossAttention_39719857553752
// MI455X (gfx1250) — compile-verified, matrix operand order checked by reading, not run on hardware
//
#include <hip/hip_runtime.h>

// ---------------------------------------------------------------------------
// CDNA5 (gfx1250) cross-attention: bf16 WMMA pipeline, TDM-fed LDS double
// buffering.  B=2, TQ=TK=2048, D=1024, H=16, HD=64 (mask read from input)
// ---------------------------------------------------------------------------

#define BATCH 2
#define TQ    2048
#define TK    2048
#define DMODEL 1024
#define NHEAD 16
#define HD    64
#define SCALE 0.125f   // 1/sqrt(64)

typedef __attribute__((ext_vector_type(16))) __bf16 v16bf;
typedef __attribute__((ext_vector_type(8)))  __bf16 v8bf;
typedef __attribute__((ext_vector_type(8)))  float  v8f;
typedef __attribute__((ext_vector_type(4)))  unsigned v4u;
typedef __attribute__((ext_vector_type(8)))  int      v8i;
typedef __attribute__((ext_vector_type(4)))  int      v4i;

union Frag16 { v16bf v; v8bf h[2]; };
union Pack8  { uint4 u; __bf16 b[8]; };

// ---- CDNA5 feature detection (compile-safe fallbacks) ----------------------
#if defined(__gfx1250__) && __has_builtin(__builtin_amdgcn_tensor_load_to_lds) && \
    __has_builtin(__builtin_amdgcn_s_wait_tensorcnt)
#define USE_TDM 1
#if __has_include(<hip/amd_detail/amd_gfx1250_TDM.h>)
#define TDM_SIX_ARGS 1   // amdgpu-toolchain / therock-10.0 headers => 6-arg builtin
#endif
#endif

#if __has_builtin(__builtin_amdgcn_global_load_async_to_lds_b128) && \
    __has_builtin(__builtin_amdgcn_s_wait_asynccnt)
#define USE_ASYNC_LDS 1
#endif

#define AS1 __attribute__((address_space(1)))
#define AS3 __attribute__((address_space(3)))

__device__ __forceinline__ __bf16 f2bf(float f) {
  unsigned u = __builtin_bit_cast(unsigned, f);
  u = u + 0x7FFFu + ((u >> 16) & 1u);           // round-to-nearest-even
  unsigned short s = (unsigned short)(u >> 16);
  return __builtin_bit_cast(__bf16, s);
}

__device__ __forceinline__ v8f wmma_bf16(v16bf a, v16bf b, v8f c) {
  return __builtin_amdgcn_wmma_f32_16x16x32_bf16(
      /*neg_a=*/false, a, /*neg_b=*/false, b,
      /*c_mod=*/(short)0, c, /*reuse_a=*/false, /*reuse_b=*/false);
}

#if defined(USE_TDM)
// Issue one TDM descriptor: copy a tile_d1-row x tile_d0-col bf16 tile
// (row stride = stride0 elements, row-major) from global into LDS, with LDS
// row padding producing a conflict-skewed stride.
__device__ __forceinline__ void tdm_load_2d(const __bf16* gsrc,
                                            unsigned lds_byte_addr,
                                            unsigned tile_d0, unsigned tile_d1,
                                            unsigned long long stride0,
                                            unsigned tensor_d0, unsigned tensor_d1,
                                            unsigned pad_int_code, unsigned pad_amt_code) {
  unsigned long long ga = (unsigned long long)(size_t)gsrc;
  v4u g0;
  g0.x = 1u;                                             // count=1, user mode
  g0.y = lds_byte_addr;                                  // lds_addr
  g0.z = (unsigned)(ga & 0xFFFFFFFFull);                 // global_addr[31:0]
  g0.w = (unsigned)((ga >> 32) & 0x1FFFFFFull) | (2u << 30);  // addr[56:32] | type=2
  v8i g1;
  g1[0] = (int)((1u << 16)                 // data_size = 2 bytes (bf16)
              | (1u << 20)                 // pad_enable
              | (pad_int_code << 22)       // pad_interval
              | (pad_amt_code << 25));     // pad_amount
  g1[1] = (int)((tensor_d0 & 0xFFFFu) << 16);                  // tensor_dim0[15:0]
  g1[2] = (int)((tensor_d0 >> 16) | ((tensor_d1 & 0xFFFFu) << 16));
  g1[3] = (int)((tensor_d1 >> 16) | (tile_d0 << 16));          // | tile_dim0
  g1[4] = (int)(tile_d1);                                      // tile_dim1, tile_dim2=0
  g1[5] = (int)(stride0 & 0xFFFFFFFFull);                      // stride0[31:0]
  g1[6] = (int)((stride0 >> 32) & 0xFFFFull);                  // stride0 hi | stride1 lo
  g1[7] = 0;
  v4i gz = {0, 0, 0, 0};
#if defined(TDM_SIX_ARGS)
  v8i gz8 = {0, 0, 0, 0, 0, 0, 0, 0};
  __builtin_amdgcn_tensor_load_to_lds(g0, g1, gz, gz, gz8, 0);
#else
  __builtin_amdgcn_tensor_load_to_lds(g0, g1, gz, gz, 0);
#endif
}
#endif

// ---------------------------------------------------------------------------
// fp32 -> bf16 down-convert (vectorized, grid-stride)
// ---------------------------------------------------------------------------
__global__ void cvt_f32_bf16(const float* __restrict__ s, __bf16* __restrict__ d, int n) {
  int i = (blockIdx.x * blockDim.x + threadIdx.x) * 4;
  int stride = gridDim.x * blockDim.x * 4;
  for (; i + 3 < n; i += stride) {
    float4 f = *(const float4*)(s + i);
    d[i + 0] = f2bf(f.x);
    d[i + 1] = f2bf(f.y);
    d[i + 2] = f2bf(f.z);
    d[i + 3] = f2bf(f.w);
  }
}

// ---------------------------------------------------------------------------
// Generic bf16 WMMA GEMM:  C[M,N] = A[M,K] * W[K,N] + bias[N]
//   block tile 128x128, BK=32, 8 waves (4x2), 32x64 per wave.
//   Double-buffered LDS: tile i+1 staged (A via TDM, B via transposing
//   scatter) while tile i is consumed; one barrier per iteration.
//   MODE 0: Q projection  -> outK1 as [B,H,TQ,HD] bf16
//   MODE 1: KV projection -> outK1=K [B,H,TK,HD], outK2=V [B,H,TK,HD] bf16
//   MODE 2: O projection  -> outF fp32 row-major [M,N]
// ---------------------------------------------------------------------------
#define LDT 40     // GEMM LDS row stride in elements (80 B, 16 B-aligned, skewed)
#define ATILE (128 * LDT)

template <int MODE>
__global__ __launch_bounds__(256) void wmma_gemm(
    const __bf16* __restrict__ A, const __bf16* __restrict__ W,
    const float* __restrict__ bias,
    __bf16* __restrict__ outK1, __bf16* __restrict__ outK2,
    float* __restrict__ outF,
    int M, int N, int K)
{
  __shared__ __bf16 As[2 * ATILE];
  __shared__ __bf16 Bs[2 * ATILE];

  const int t     = threadIdx.x;
  const int lane  = t & 31;
  const int wid   = t >> 5;
  const int waveM = wid >> 1;          // 0..3
  const int waveN = wid & 1;           // 0..1
  const int m0    = blockIdx.x * 128;
  const int n0    = blockIdx.y * 128;
  const int col   = lane & 15;
  const int khalf = lane >> 4;

  const int arow = t >> 1;             // 0..127
  const int ach  = (t & 1) * 16;       // 0 / 16
  const int brow = t >> 3;             // 0..31 (k)
  const int bnc  = (t & 7) * 16;       // 0..112 (n)

  // --- staging helpers ---
  const auto stage_a = [&](int kk, int buf) {
    __bf16* dst = &As[buf * ATILE];
#if defined(USE_TDM)
    if (wid == 0) {
      // pad_interval code 3 = 16 DWORDs (one 32-elem row); pad 4 DWORDs -> LDT=40
      tdm_load_2d(A + (size_t)m0 * K + kk,
                  (unsigned)(unsigned long long)(void*)dst, 32, 128,
                  (unsigned long long)K, (unsigned)K, (unsigned)M, 3, 3);
    }
#else
    const __bf16* asrc = A + (size_t)(m0 + arow) * K + kk + ach;
#if defined(USE_ASYNC_LDS)
    __builtin_amdgcn_global_load_async_to_lds_b128(
        (const AS1 void*)asrc, (AS3 void*)&dst[arow * LDT + ach], 0, 0);
    __builtin_amdgcn_global_load_async_to_lds_b128(
        (const AS1 void*)(asrc + 8), (AS3 void*)&dst[arow * LDT + ach + 8], 0, 0);
#else
    *(uint4*)&dst[arow * LDT + ach]     = *(const uint4*)(asrc);
    *(uint4*)&dst[arow * LDT + ach + 8] = *(const uint4*)(asrc + 8);
#endif
#endif
  };
  const auto stage_b = [&](int kk, int buf) {
    __bf16* dst = &Bs[buf * ATILE];
    const __bf16* bsrc = W + (size_t)(kk + brow) * N + n0 + bnc;
    if (kk + 32 < K) __builtin_prefetch(bsrc + (size_t)32 * N, 0, 1);
    Pack8 d0, d1;
    d0.u = *(const uint4*)(bsrc);
    d1.u = *(const uint4*)(bsrc + 8);
#pragma unroll
    for (int e = 0; e < 8; ++e) {
      dst[(bnc + e)     * LDT + brow] = d0.b[e];
      dst[(bnc + 8 + e) * LDT + brow] = d1.b[e];
    }
  };
  const auto wait_stage = [&]() {
#if defined(USE_TDM)
    if (wid == 0) __builtin_amdgcn_s_wait_tensorcnt(0);
#elif defined(USE_ASYNC_LDS)
    __builtin_amdgcn_s_wait_asynccnt(0);
#endif
  };

  v8f acc[2][4];
#pragma unroll
  for (int i = 0; i < 2; ++i)
#pragma unroll
    for (int j = 0; j < 4; ++j)
#pragma unroll
      for (int r = 0; r < 8; ++r) acc[i][j][r] = 0.0f;

  const int NK = K >> 5;               // k-steps of 32

  // --- prologue: stage tile 0 into buffer 0 ---
  stage_a(0, 0);
  stage_b(0, 0);
  wait_stage();
  __syncthreads();

  for (int i = 0; i < NK; ++i) {
    const int cur = i & 1;
    const int nxt = cur ^ 1;
    if (i + 1 < NK) {                   // overlap: stage tile i+1
      stage_a((i + 1) << 5, nxt);
      stage_b((i + 1) << 5, nxt);
    }

    const __bf16* Ac = &As[cur * ATILE];
    const __bf16* Bc = &Bs[cur * ATILE];
    Frag16 af[2], bfr[4];
#pragma unroll
    for (int mt = 0; mt < 2; ++mt) {
      const __bf16* p = &Ac[(waveM * 32 + mt * 16 + col) * LDT + khalf * 8];
      af[mt].h[0] = *(const v8bf*)(p);
      af[mt].h[1] = *(const v8bf*)(p + 16);
    }
#pragma unroll
    for (int nt = 0; nt < 4; ++nt) {
      const __bf16* p = &Bc[(waveN * 64 + nt * 16 + col) * LDT + khalf * 8];
      bfr[nt].h[0] = *(const v8bf*)(p);
      bfr[nt].h[1] = *(const v8bf*)(p + 16);
    }
#pragma unroll
    for (int mt = 0; mt < 2; ++mt)
#pragma unroll
      for (int nt = 0; nt < 4; ++nt)
        acc[mt][nt] = wmma_bf16(af[mt].v, bfr[nt].v, acc[mt][nt]);

    if (i + 1 < NK) wait_stage();       // DMA for i+1 must land before next iter
    __syncthreads();
  }

  // --- epilogue (MODE is a compile-time constant) ---
#pragma unroll
  for (int mt = 0; mt < 2; ++mt) {
#pragma unroll
    for (int nt = 0; nt < 4; ++nt) {
#pragma unroll
      for (int r = 0; r < 8; ++r) {
        int row = m0 + waveM * 32 + mt * 16 + khalf * 8 + r;
        int cn  = n0 + waveN * 64 + nt * 16 + col;
        float v = acc[mt][nt][r] + bias[cn];
        if (MODE == 0) {
          int bb = row >> 11, q = row & 2047, hh = cn >> 6, d = cn & 63;
          outK1[(((size_t)(bb * NHEAD + hh)) * TQ + q) * HD + d] = f2bf(v);
        } else if (MODE == 1) {
          int bb = row >> 11, tt = row & 2047;
          if (cn < DMODEL) {
            int hh = cn >> 6, d = cn & 63;
            outK1[(((size_t)(bb * NHEAD + hh)) * TK + tt) * HD + d] = f2bf(v);
          } else {
            int c2 = cn - DMODEL, hh = c2 >> 6, d = c2 & 63;
            outK2[(((size_t)(bb * NHEAD + hh)) * TK + tt) * HD + d] = f2bf(v);
          }
        } else {
          outF[(size_t)row * DMODEL + cn] = v;
        }
      }
    }
  }
}

// ---------------------------------------------------------------------------
// Flash attention, block-cooperative + double-buffered key tiles:
//   One block = one (b, h, 128-row q group); 8 waves take 16 q rows each.
//   Per 64-key tile: K tile (64x64 row-major, 72-elem stride) staged by the
//   TDM; V tile staged *transposed* ([hd][key]) by all threads.  Tile i+1 is
//   staged while tile i is consumed.
// ---------------------------------------------------------------------------
#define LKV 72       // key-tile LDS stride in elements (144 B = 9*16, skewed)
#define KVTILE (64 * LKV)

__global__ __launch_bounds__(256) void flash_attn(
    const __bf16* __restrict__ Q, const __bf16* __restrict__ Kk,
    const __bf16* __restrict__ Vv, const unsigned char* __restrict__ kpm,
    __bf16* __restrict__ O)
{
  __shared__ __bf16 Ks[2 * KVTILE];     // [key][hd]
  __shared__ __bf16 Vs[2 * KVTILE];     // [hd][key] (transposed)
  __shared__ __bf16 Pl[8][16 * 64];

  const int t     = threadIdx.x;
  const int lane  = t & 31;
  const int wid   = t >> 5;
  const int bx    = blockIdx.x;          // 0 .. B*H*(TQ/128)-1
  const int b     = bx >> 8;
  const int h     = (bx >> 4) & 15;
  const int q0    = (bx & 15) * 128;
  const int col   = lane & 15;
  const int khalf = lane >> 4;

  const __bf16* Qh = Q  + (((size_t)(b * NHEAD + h)) * TQ + q0 + wid * 16) * HD;
  const __bf16* Kh = Kk + ((size_t)(b * NHEAD + h)) * TK * HD;
  const __bf16* Vh = Vv + ((size_t)(b * NHEAD + h)) * TK * HD;

  // staging indices: thread t copies 16 hd-contiguous elems of one key row
  const int vkey = t >> 2;               // 0..63
  const int vhc  = (t & 3) * 16;         // 0..48

  const auto stage_kv = [&](int kk, int buf) {
    __bf16* kd = &Ks[buf * KVTILE];
    __bf16* vd = &Vs[buf * KVTILE];
#if defined(USE_TDM)
    if (wid == 0) {
      // pad_interval code 4 = 32 DWORDs (one 64-elem row); pad 4 DWORDs -> LKV=72
      tdm_load_2d(Kh + (size_t)kk * HD,
                  (unsigned)(unsigned long long)(void*)kd, 64, 64,
                  (unsigned long long)HD, (unsigned)HD, (unsigned)TK, 4, 3);
    }
#else
    {
      const __bf16* ksrc = Kh + (size_t)(kk + vkey) * HD + vhc;
#if defined(USE_ASYNC_LDS)
      __builtin_amdgcn_global_load_async_to_lds_b128(
          (const AS1 void*)ksrc, (AS3 void*)&kd[vkey * LKV + vhc], 0, 0);
      __builtin_amdgcn_global_load_async_to_lds_b128(
          (const AS1 void*)(ksrc + 8), (AS3 void*)&kd[vkey * LKV + vhc + 8], 0, 0);
#else
      *(uint4*)&kd[vkey * LKV + vhc]     = *(const uint4*)(ksrc);
      *(uint4*)&kd[vkey * LKV + vhc + 8] = *(const uint4*)(ksrc + 8);
#endif
    }
#endif
    // V tile transposed: Vs[hd][key]
    const __bf16* vsrc = Vh + (size_t)(kk + vkey) * HD + vhc;
    Pack8 d0, d1;
    d0.u = *(const uint4*)(vsrc);
    d1.u = *(const uint4*)(vsrc + 8);
#pragma unroll
    for (int e = 0; e < 8; ++e) {
      vd[(vhc + e)     * LKV + vkey] = d0.b[e];
      vd[(vhc + 8 + e) * LKV + vkey] = d1.b[e];
    }
  };
  const auto wait_stage = [&]() {
#if defined(USE_TDM)
    if (wid == 0) __builtin_amdgcn_s_wait_tensorcnt(0);
#elif defined(USE_ASYNC_LDS)
    __builtin_amdgcn_s_wait_asynccnt(0);
#endif
  };

  // Q fragments (A-layout), constant over the key loop. 2 k-steps of 32 over HD=64
  Frag16 qf[2];
#pragma unroll
  for (int st = 0; st < 2; ++st) {
    const __bf16* p = Qh + (size_t)col * HD + st * 32 + khalf * 8;
    qf[st].h[0] = *(const v8bf*)(p);
    qf[st].h[1] = *(const v8bf*)(p + 16);
  }

  v8f   o_[4];
  float mrun[8], lp[8];
#pragma unroll
  for (int j = 0; j < 4; ++j)
#pragma unroll
    for (int r = 0; r < 8; ++r) o_[j][r] = 0.0f;
#pragma unroll
  for (int r = 0; r < 8; ++r) { mrun[r] = -1e30f; lp[r] = 0.0f; }

  const int NT = TK / 64;               // 32 key tiles

  // --- prologue: stage tile 0 into buffer 0 ---
  stage_kv(0, 0);
  wait_stage();
  __syncthreads();

  for (int i = 0; i < NT; ++i) {
    const int cur = i & 1;
    const int nxt = cur ^ 1;
    const int k0  = i * 64;
    if (i + 1 < NT) stage_kv(k0 + 64, nxt);    // overlap with compute below

    const __bf16* Kc = &Ks[cur * KVTILE];
    const __bf16* Vc = &Vs[cur * KVTILE];

    // ---- S = Q K^T (16 x 64), K fragments from LDS ----
    v8f s_[4];
#pragma unroll
    for (int j = 0; j < 4; ++j)
#pragma unroll
      for (int r = 0; r < 8; ++r) s_[j][r] = 0.0f;

#pragma unroll
    for (int j = 0; j < 4; ++j) {
#pragma unroll
      for (int st = 0; st < 2; ++st) {
        Frag16 kf;
        const __bf16* p = &Kc[(j * 16 + col) * LKV + st * 32 + khalf * 8];
        kf.h[0] = *(const v8bf*)(p);
        kf.h[1] = *(const v8bf*)(p + 16);
        s_[j] = wmma_bf16(qf[st].v, kf.v, s_[j]);
      }
    }

    // ---- scale + key-padding mask + online softmax ----
    float pr[4][8], mnew[8];
#pragma unroll
    for (int r = 0; r < 8; ++r) mnew[r] = mrun[r];
#pragma unroll
    for (int j = 0; j < 4; ++j) {
      const int key = k0 + j * 16 + col;
      const float mb = kpm[b * TK + key] ? -1e30f : 0.0f;
#pragma unroll
      for (int r = 0; r < 8; ++r) {
        float v = s_[j][r] * SCALE + mb;
        pr[j][r] = v;
        mnew[r] = fmaxf(mnew[r], v);
      }
    }
    // row-max across the 16-lane half (rows live in fixed half-wave groups)
#pragma unroll
    for (int msk = 1; msk < 16; msk <<= 1)
#pragma unroll
      for (int r = 0; r < 8; ++r)
        mnew[r] = fmaxf(mnew[r], __shfl_xor(mnew[r], msk));

    float alpha[8];
#pragma unroll
    for (int r = 0; r < 8; ++r) {
      alpha[r] = __expf(mrun[r] - mnew[r]);
      mrun[r] = mnew[r];
    }
#pragma unroll
    for (int j = 0; j < 4; ++j)
#pragma unroll
      for (int r = 0; r < 8; ++r)
        pr[j][r] = __expf(pr[j][r] - mnew[r]);
#pragma unroll
    for (int r = 0; r < 8; ++r)
      lp[r] = lp[r] * alpha[r] + (pr[0][r] + pr[1][r] + pr[2][r] + pr[3][r]);
#pragma unroll
    for (int j = 0; j < 4; ++j)
#pragma unroll
      for (int r = 0; r < 8; ++r) o_[j][r] *= alpha[r];

    // ---- stage P (C-layout -> row-major LDS -> A-layout) ----
#pragma unroll
    for (int j = 0; j < 4; ++j)
#pragma unroll
      for (int r = 0; r < 8; ++r)
        Pl[wid][(khalf * 8 + r) * 64 + j * 16 + col] = f2bf(pr[j][r]);
    __syncthreads();

    // ---- O += P V  (contraction over 64 keys = 2 k-steps of 32) ----
#pragma unroll
    for (int st = 0; st < 2; ++st) {
      Frag16 pf;
      const __bf16* pp = &Pl[wid][col * 64 + st * 32 + khalf * 8];
      pf.h[0] = *(const v8bf*)(pp);
      pf.h[1] = *(const v8bf*)(pp + 16);
#pragma unroll
      for (int j = 0; j < 4; ++j) {
        Frag16 vf;
        const __bf16* vp = &Vc[(j * 16 + col) * LKV + st * 32 + khalf * 8];
        vf.h[0] = *(const v8bf*)(vp);
        vf.h[1] = *(const v8bf*)(vp + 16);
        o_[j] = wmma_bf16(pf.v, vf.v, o_[j]);
      }
    }

    if (i + 1 < NT) wait_stage();       // DMA for i+1 must land before next iter
    __syncthreads();
  }

  // ---- normalize and store O as [B,TQ,D] with heads interleaved back ----
#pragma unroll
  for (int msk = 1; msk < 16; msk <<= 1)
#pragma unroll
    for (int r = 0; r < 8; ++r) lp[r] += __shfl_xor(lp[r], msk);

#pragma unroll
  for (int r = 0; r < 8; ++r) lp[r] = 1.0f / lp[r];

#pragma unroll
  for (int j = 0; j < 4; ++j)
#pragma unroll
    for (int r = 0; r < 8; ++r) {
      const int row = q0 + wid * 16 + khalf * 8 + r;
      const float v = o_[j][r] * lp[r];
      O[((size_t)(b * TQ + row)) * DMODEL + h * HD + j * 16 + col] = f2bf(v);
    }
}

// ---------------------------------------------------------------------------
// Host-side launcher
// ---------------------------------------------------------------------------
extern "C" void kernel_launch(void* const* d_in, const int* in_sizes, int n_in,
                              void* d_out, int out_size, void* d_ws, size_t ws_size,
                              hipStream_t stream) {
  (void)in_sizes; (void)n_in; (void)out_size; (void)ws_size;

  const float*         q    = (const float*)d_in[0];
  const float*         kv   = (const float*)d_in[1];
  const unsigned char* mask = (const unsigned char*)d_in[2];
  const float*         Wq   = (const float*)d_in[3];
  const float*         bq   = (const float*)d_in[4];
  const float*         Wkv  = (const float*)d_in[5];
  const float*         bkv  = (const float*)d_in[6];
  const float*         Wo   = (const float*)d_in[7];
  const float*         bo   = (const float*)d_in[8];
  float*               out  = (float*)d_out;

  // workspace partition (bf16 tensors)
  char* p = (char*)d_ws;
  __bf16* qbf   = (__bf16*)p;                 p += (size_t)BATCH * TQ * DMODEL * 2;      // 8 MB
  __bf16* kvbf  = (__bf16*)p;                 p += (size_t)BATCH * TK * DMODEL * 2;      // 8 MB
  __bf16* wqb   = (__bf16*)p;                 p += (size_t)DMODEL * DMODEL * 2;          // 2 MB
  __bf16* wkvb  = (__bf16*)p;                 p += (size_t)DMODEL * 2 * DMODEL * 2;      // 4 MB
  __bf16* wob   = (__bf16*)p;                 p += (size_t)DMODEL * DMODEL * 2;          // 2 MB
  __bf16* Qh    = (__bf16*)p;                 p += (size_t)BATCH * TQ * DMODEL * 2;      // 8 MB
  __bf16* Kh    = (__bf16*)p;                 p += (size_t)BATCH * TK * DMODEL * 2;      // 8 MB
  __bf16* Vh    = (__bf16*)p;                 p += (size_t)BATCH * TK * DMODEL * 2;      // 8 MB
  __bf16* Oh    = (__bf16*)p;                 /* 8 MB */

  const int nQ   = BATCH * TQ * DMODEL;    // 4 Mi
  const int nKV  = BATCH * TK * DMODEL;
  const int nWq  = DMODEL * DMODEL;
  const int nWkv = DMODEL * 2 * DMODEL;

  cvt_f32_bf16<<<4096, 256, 0, stream>>>(q,   qbf,  nQ);
  cvt_f32_bf16<<<4096, 256, 0, stream>>>(kv,  kvbf, nKV);
  cvt_f32_bf16<<<1024, 256, 0, stream>>>(Wq,  wqb,  nWq);
  cvt_f32_bf16<<<2048, 256, 0, stream>>>(Wkv, wkvb, nWkv);
  cvt_f32_bf16<<<1024, 256, 0, stream>>>(Wo,  wob,  nWq);

  // Q projection: [4096,1024] = qbf * Wq + bq  -> Qh [B,H,TQ,HD]
  wmma_gemm<0><<<dim3(32, 8), 256, 0, stream>>>(qbf, wqb, bq, Qh, nullptr, nullptr,
                                                BATCH * TQ, DMODEL, DMODEL);
  // KV projection: [4096,2048] = kvbf * Wkv + bkv -> Kh, Vh
  wmma_gemm<1><<<dim3(32, 16), 256, 0, stream>>>(kvbf, wkvb, bkv, Kh, Vh, nullptr,
                                                 BATCH * TK, 2 * DMODEL, DMODEL);
  // attention: one block per (b, h, 128 q rows) -> 512 blocks
  flash_attn<<<512, 256, 0, stream>>>(Qh, Kh, Vh, mask, Oh);
  // output projection: fp32 result
  wmma_gemm<2><<<dim3(32, 8), 256, 0, stream>>>(Oh, wob, bo, nullptr, nullptr, out,
                                                BATCH * TQ, DMODEL, DMODEL);
}
